// MultiHeadSelfAttentionRPE_58823872086116
// MI455X (gfx1250) — matrix operand order repaired and checked by reading, not compiled, not run on hardware
//
#include <hip/hip_runtime.h>
#include <hip/hip_bf16.h>

typedef __attribute__((ext_vector_type(16))) __bf16 bf16x16;
typedef __attribute__((ext_vector_type(8)))  float  f32x8;

union FragB16 { bf16x16 v; uint4 q[2]; };

#define T_SEQ   2048
#define B_BATCH 2
#define NHEADS  16
#define DMODEL  1024
#define RPE_H   32

__device__ __forceinline__ f32x8 zero8() {
  f32x8 z;
#pragma unroll
  for (int i = 0; i < 8; ++i) z[i] = 0.0f;
  return z;
}

__device__ __forceinline__ f32x8 wmma_bf16(bf16x16 a, bf16x16 b, f32x8 c) {
  return __builtin_amdgcn_wmma_f32_16x16x32_bf16(false, a, false, b, (short)0, c, false, false);
}

// ---------------- conversion / transpose ----------------

__global__ void cvt_bf16_kernel(const float* __restrict__ src, __bf16* __restrict__ dst, int n) {
  int i = blockIdx.x * blockDim.x + threadIdx.x;
  if (i < n) dst[i] = (__bf16)src[i];
}

// src [rows][cols] f32 -> dst [cols][rows] bf16
__global__ void transpose_bf16_kernel(const float* __restrict__ src, __bf16* __restrict__ dst,
                                      int rows, int cols) {
  int i = blockIdx.x * blockDim.x + threadIdx.x;
  if (i < rows * cols) {
    int r = i / cols, c = i % cols;
    dst[(size_t)c * rows + r] = (__bf16)src[i];
  }
}

// ---------------- RPE bias table: bias[h][t-s+2047], 16 x 4095 ----------------

__global__ void rpe_table_kernel(const float* __restrict__ w1, const float* __restrict__ b1,
                                 const float* __restrict__ w2, const float* __restrict__ b2,
                                 float* __restrict__ tab) {
  int r = blockIdx.x * blockDim.x + threadIdx.x;
  if (r >= 2 * T_SEQ - 1) return;
  float rel = (float)(r - (T_SEQ - 1));
  float v = rel > 0.0f ? log1pf(rel) : (rel < 0.0f ? -log1pf(-rel) : 0.0f);
  float acc[NHEADS];
#pragma unroll
  for (int h = 0; h < NHEADS; ++h) acc[h] = b2[h];
  for (int j = 0; j < RPE_H; ++j) {
    float hj = fmaxf(v * w1[j] + b1[j], 0.0f);
#pragma unroll
    for (int h = 0; h < NHEADS; ++h) acc[h] += hj * w2[j * NHEADS + h];
  }
#pragma unroll
  for (int h = 0; h < NHEADS; ++h) tab[h * 4096 + r] = acc[h];
}

// ---------------- GEMM: C[M,N] f32 = A[M,K] bf16 * Bt[N,K] bf16 + bias[N] ------
// 256 threads = 8 waves; block tile 128x256; wave tile 64x64 (4x4 WMMA accums).

__global__ __launch_bounds__(256) void gemm_bf16_kernel(
    const __bf16* __restrict__ A, const __bf16* __restrict__ Bt,
    const float* __restrict__ bias, float* __restrict__ C,
    int M, int N, int K) {
  const int lane = threadIdx.x & 31, w = threadIdx.x >> 5;
  const int lm = lane & 15, g = lane >> 4;
  const int mw = blockIdx.y * 128 + (w >> 2) * 64;
  const int nw = blockIdx.x * 256 + (w & 3) * 64;

  f32x8 acc[4][4];
#pragma unroll
  for (int i = 0; i < 4; ++i)
#pragma unroll
    for (int j = 0; j < 4; ++j) acc[i][j] = zero8();

  for (int k0 = 0; k0 < K; k0 += 32) {
    FragB16 a[4], b[4];
#pragma unroll
    for (int i = 0; i < 4; ++i) {
      const __bf16* p = A + (size_t)(mw + 16 * i + lm) * K + k0;
      a[i].q[0] = *(const uint4*)(p + 16 * g);       // K = 16g..16g+7
      a[i].q[1] = *(const uint4*)(p + 16 * g + 8);   // K = 16g+8..16g+15
      __builtin_prefetch(p + 32, 0, 0);
    }
#pragma unroll
    for (int j = 0; j < 4; ++j) {
      const __bf16* p = Bt + (size_t)(nw + 16 * j + lm) * K + k0 + 16 * g;
      b[j].q[0] = *(const uint4*)p;                  // K = 16g..16g+7
      b[j].q[1] = *(const uint4*)(p + 8);            // K = 16g+8..16g+15
      __builtin_prefetch(p + 32, 0, 0);
    }
#pragma unroll
    for (int i = 0; i < 4; ++i)
#pragma unroll
      for (int j = 0; j < 4; ++j)
        acc[i][j] = wmma_bf16(a[i].v, b[j].v, acc[i][j]);
  }

#pragma unroll
  for (int i = 0; i < 4; ++i)
#pragma unroll
    for (int j = 0; j < 4; ++j) {
      const int col = nw + 16 * j + lm;
      const float bn = bias[col];
#pragma unroll
      for (int r = 0; r < 8; ++r) {
        const int row = mw + 16 * i + r + 8 * g;
        C[(size_t)row * N + col] = acc[i][j][r] + bn;
      }
    }
}

// ---------------- split qkv f32 [B*T][3072] -> q/k bf16 [b][h][t][d], vT [b][h][d][t]

__global__ void qkv_split_kernel(const float* __restrict__ qkv,
                                 __bf16* __restrict__ qb, __bf16* __restrict__ kb,
                                 __bf16* __restrict__ vt) {
  long i = (long)blockIdx.x * blockDim.x + threadIdx.x;
  const long total = (long)B_BATCH * T_SEQ * 3 * DMODEL;
  if (i >= total) return;
  int col = (int)(i % (3 * DMODEL));
  long row = i / (3 * DMODEL);
  int b = (int)(row / T_SEQ), t = (int)(row % T_SEQ);
  int which = col / DMODEL, cc = col % DMODEL;
  int h = cc >> 6, d = cc & 63;
  float v = qkv[i];
  long bh = (long)b * NHEADS + h;
  if (which == 0)      qb[(bh * T_SEQ + t) * 64 + d] = (__bf16)(v * 0.125f);  // 1/sqrt(64)
  else if (which == 1) kb[(bh * T_SEQ + t) * 64 + d] = (__bf16)v;
  else                 vt[(bh * 64 + d) * T_SEQ + t] = (__bf16)v;
}

// ---------------- flash attention with RPE bias -------------------------------
// grid = B*H*(T/128); 8 waves/block; each wave: 16 query rows, online softmax.

__global__ __launch_bounds__(256) void attn_kernel(
    const __bf16* __restrict__ qb, const __bf16* __restrict__ kb,
    const __bf16* __restrict__ vt, const float* __restrict__ biastab,
    __bf16* __restrict__ attout) {
  __shared__ float biasS[4096];
  __shared__ __align__(16) __bf16 Pbuf[8][16][32];

  const int bid = blockIdx.x;
  const int b = bid / (NHEADS * 16);
  const int rem = bid % (NHEADS * 16);
  const int h = rem / 16;
  const int tblk = (rem % 16) * 128;

  for (int idx = threadIdx.x; idx < 2 * T_SEQ - 1; idx += 256)
    biasS[idx] = biastab[h * 4096 + idx];
  __syncthreads();

  const int w = threadIdx.x >> 5, lane = threadIdx.x & 31;
  const int lm = lane & 15, g = lane >> 4;
  const int t0 = tblk + w * 16;
  const long bh = (long)b * NHEADS + h;

  // Q fragments (A-operand), d in [0,32) and [32,64)
  FragB16 qa[2];
  {
    const __bf16* qp = qb + (bh * T_SEQ + t0 + lm) * 64;
#pragma unroll
    for (int j = 0; j < 2; ++j) {
      qa[j].q[0] = *(const uint4*)(qp + 32 * j + 16 * g);
      qa[j].q[1] = *(const uint4*)(qp + 32 * j + 16 * g + 8);
    }
  }

  f32x8 acc[4];
#pragma unroll
  for (int d4 = 0; d4 < 4; ++d4) acc[d4] = zero8();
  float mrun[8], lrun[8];
#pragma unroll
  for (int r = 0; r < 8; ++r) { mrun[r] = -3.0e38f; lrun[r] = 0.0f; }

  for (int s0 = 0; s0 < T_SEQ; s0 += 32) {
    // ---- S = Q*K^T for 32 keys, two 16x16 C-tiles
    f32x8 S[2];
#pragma unroll
    for (int c = 0; c < 2; ++c) {
      const __bf16* kp = kb + (bh * T_SEQ + s0 + 16 * c + lm) * 64;
      FragB16 kf0, kf1;
      kf0.q[0] = *(const uint4*)(kp + 16 * g);
      kf0.q[1] = *(const uint4*)(kp + 16 * g + 8);
      kf1.q[0] = *(const uint4*)(kp + 32 + 16 * g);
      kf1.q[1] = *(const uint4*)(kp + 32 + 16 * g + 8);
      f32x8 z = zero8();
      z = wmma_bf16(qa[0].v, kf0.v, z);
      S[c] = wmma_bf16(qa[1].v, kf1.v, z);
    }

    // ---- bias add (translation-invariant table in LDS)
#pragma unroll
    for (int r = 0; r < 8; ++r) {
      const int t = t0 + r + 8 * g;
      S[0][r] += biasS[t - (s0 + lm) + (T_SEQ - 1)];
      S[1][r] += biasS[t - (s0 + 16 + lm) + (T_SEQ - 1)];
    }

    // ---- online softmax (rows live across 16-lane halves)
#pragma unroll
    for (int r = 0; r < 8; ++r) {
      float x = fmaxf(S[0][r], S[1][r]);
      x = fmaxf(x, __shfl_xor(x, 1, 32));
      x = fmaxf(x, __shfl_xor(x, 2, 32));
      x = fmaxf(x, __shfl_xor(x, 4, 32));
      x = fmaxf(x, __shfl_xor(x, 8, 32));
      const float mn = fmaxf(mrun[r], x);
      const float alpha = __expf(mrun[r] - mn);
      mrun[r] = mn;
      const float p0 = __expf(S[0][r] - mn);
      const float p1 = __expf(S[1][r] - mn);
      float rs = p0 + p1;
      rs += __shfl_xor(rs, 1, 32);
      rs += __shfl_xor(rs, 2, 32);
      rs += __shfl_xor(rs, 4, 32);
      rs += __shfl_xor(rs, 8, 32);
      lrun[r] = lrun[r] * alpha + rs;
      Pbuf[w][r + 8 * g][lm]      = (__bf16)p0;
      Pbuf[w][r + 8 * g][lm + 16] = (__bf16)p1;
#pragma unroll
      for (int d4 = 0; d4 < 4; ++d4) acc[d4][r] *= alpha;
    }

    // wave-synchronous LDS transpose: wait for our own ds stores, then reload
    asm volatile("s_wait_dscnt 0" ::: "memory");
    FragB16 pf;
    pf.q[0] = *(const uint4*)(&Pbuf[w][lm][16 * g]);
    pf.q[1] = *(const uint4*)(&Pbuf[w][lm][16 * g + 8]);

    // ---- acc += P * V (V^T rows are contiguous in s)
#pragma unroll
    for (int d4 = 0; d4 < 4; ++d4) {
      const __bf16* vp = vt + (bh * 64 + 16 * d4 + lm) * T_SEQ + s0 + 16 * g;
      FragB16 vf;
      vf.q[0] = *(const uint4*)vp;
      vf.q[1] = *(const uint4*)(vp + 8);
      acc[d4] = wmma_bf16(pf.v, vf.v, acc[d4]);
    }
  }

  // ---- epilogue: normalize, write [B,T,C] bf16 for the output projection
#pragma unroll
  for (int r = 0; r < 8; ++r) {
    const float inv = 1.0f / lrun[r];
    const int t = t0 + r + 8 * g;
    const long orow = ((long)b * T_SEQ + t) * DMODEL + h * 64;
#pragma unroll
    for (int d4 = 0; d4 < 4; ++d4)
      attout[orow + 16 * d4 + lm] = (__bf16)(acc[d4][r] * inv);
  }
}

// ---------------- launch -------------------------------------------------------

extern "C" void kernel_launch(void* const* d_in, const int* in_sizes, int n_in,
                              void* d_out, int out_size, void* d_ws, size_t ws_size,
                              hipStream_t stream) {
  const float* x     = (const float*)d_in[0];
  const float* w_qkv = (const float*)d_in[1];
  const float* b_qkv = (const float*)d_in[2];
  const float* w_out = (const float*)d_in[3];
  const float* b_out = (const float*)d_in[4];
  const float* w1    = (const float*)d_in[5];
  const float* b1    = (const float*)d_in[6];
  const float* w2    = (const float*)d_in[7];
  const float* b2    = (const float*)d_in[8];
  float* out = (float*)d_out;
  (void)in_sizes; (void)n_in; (void)out_size; (void)ws_size;

  char* ws = (char*)d_ws;
  size_t off = 0;
  auto alloc = [&](size_t bytes) { void* p = ws + off; off += (bytes + 255) & ~(size_t)255; return p; };

  const size_t MT = (size_t)B_BATCH * T_SEQ;             // 4096 rows
  __bf16* xb     = (__bf16*)alloc(MT * DMODEL * 2);      // 8 MB
  __bf16* wqkvT  = (__bf16*)alloc((size_t)3 * DMODEL * DMODEL * 2); // 6 MB
  __bf16* woutT  = (__bf16*)alloc((size_t)DMODEL * DMODEL * 2);     // 2 MB
  float*  qkv    = (float*) alloc(MT * 3 * DMODEL * 4);  // 48 MB
  __bf16* qb     = (__bf16*)alloc(MT * DMODEL * 2);      // 8 MB
  __bf16* kbuf   = (__bf16*)alloc(MT * DMODEL * 2);      // 8 MB
  __bf16* vt     = (__bf16*)alloc(MT * DMODEL * 2);      // 8 MB
  __bf16* attout = (__bf16*)alloc(MT * DMODEL * 2);      // 8 MB
  float*  btab   = (float*) alloc((size_t)NHEADS * 4096 * 4); // 256 KB

  const int nx = (int)(MT * DMODEL);
  cvt_bf16_kernel<<<(nx + 255) / 256, 256, 0, stream>>>(x, xb, nx);
  transpose_bf16_kernel<<<(DMODEL * 3 * DMODEL + 255) / 256, 256, 0, stream>>>(w_qkv, wqkvT, DMODEL, 3 * DMODEL);
  transpose_bf16_kernel<<<(DMODEL * DMODEL + 255) / 256, 256, 0, stream>>>(w_out, woutT, DMODEL, DMODEL);
  rpe_table_kernel<<<(2 * T_SEQ - 1 + 63) / 64, 64, 0, stream>>>(w1, b1, w2, b2, btab);

  gemm_bf16_kernel<<<dim3(3 * DMODEL / 256, (int)MT / 128), 256, 0, stream>>>(
      xb, wqkvT, b_qkv, qkv, (int)MT, 3 * DMODEL, DMODEL);

  const long nsplit = (long)MT * 3 * DMODEL;
  qkv_split_kernel<<<(int)((nsplit + 255) / 256), 256, 0, stream>>>(qkv, qb, kbuf, vt);

  attn_kernel<<<B_BATCH * NHEADS * (T_SEQ / 128), 256, 0, stream>>>(qb, kbuf, vt, btab, attout);

  gemm_bf16_kernel<<<dim3(DMODEL / 256, (int)MT / 128), 256, 0, stream>>>(
      attout, woutT, b_out, out, (int)MT, DMODEL, DMODEL);
}
